// PainHead_71476845740771
// MI455X (gfx1250) — matrix operand order repaired and checked by reading, not compiled, not run on hardware
//
#include <hip/hip_runtime.h>
#include <hip/hip_bf16.h>
#include <cmath>

// Problem constants (from reference)
#define N_TOK  8192
#define D_IN   600
#define D_HID  2048
#define KTAPS  9
#define BN_EPS 1e-5f

// GEMM tiling
#define BM 64          // rows per block
#define BN 128         // cols per block
#define KC 64          // K-depth staged per barrier period (2 WMMA K-steps)
#define LDA (KC + 8)   // padded LDS row stride (bf16): 144B = 9*16B, bank spread

typedef __attribute__((ext_vector_type(16))) __bf16 v16bf;
typedef __attribute__((ext_vector_type(8)))  float  v8f;
typedef __attribute__((ext_vector_type(4)))  int    v4i;

typedef union { v16bf v; uint2 u2[4]; } Frag;

// CDNA5 async global->LDS path (ASYNCcnt-tracked, bypasses VGPRs).
// Builtin signature (from toolchain diagnostic): (v4i AS1*, v4i AS3*, Ii, Ii).
#if __has_builtin(__builtin_amdgcn_global_load_async_to_lds_b128) && \
    __has_builtin(__builtin_amdgcn_s_wait_asynccnt)
#define USE_ASYNC_LDS 1
typedef __attribute__((address_space(1))) v4i* gv4p;
typedef __attribute__((address_space(3))) v4i* lv4p;
__device__ __forceinline__ void async_copy16(const __bf16* gsrc, __bf16* ldst) {
  __builtin_amdgcn_global_load_async_to_lds_b128((gv4p)gsrc, (lv4p)ldst, 0, 0);
}
#else
#define USE_ASYNC_LDS 0
#endif

// ---------------------------------------------------------------------------
// Weight transpose + fp32 -> bf16 hi/lo split, with zero-padded rows:
//   w[Dout, Din, K] (fp32) -> wh/wl[K][Dout][DinP] (bf16), DinP = 64-aligned.
// Output-index mapping => coalesced writes; done once, bandwidth-trivial.
// Padding removes all tail branching from the GEMM's B staging.
// ---------------------------------------------------------------------------
__global__ void split_transpose_kernel(const float* __restrict__ w,
                                       __bf16* __restrict__ wh,
                                       __bf16* __restrict__ wl,
                                       int Dout, int Din, int DinP, int Ktaps) {
  long long o     = (long long)blockIdx.x * blockDim.x + threadIdx.x;
  long long total = (long long)Dout * DinP * Ktaps;
  if (o >= total) return;
  int d       = (int)(o % DinP);
  long long r = o / DinP;
  int j       = (int)(r % Dout);
  int k       = (int)(r / Dout);
  float x = (d < Din) ? w[((long long)j * Din + d) * Ktaps + k] : 0.f;
  __bf16 hi = (__bf16)x;
  wh[o] = hi;
  wl[o] = (__bf16)(x - (float)hi);
}

// ---------------------------------------------------------------------------
// LDS fragment loaders (8B-aligned ds_load_b64 pieces).
// A fragment (16x32 bf16): lanes<16 hold K {0..7,16..23}, lanes>=16 {8..15,24..31}.
// B fragment (32x16 bf16): same K split as A (lanes<16 K {0..7,16..23}, lanes>=16 {8..15,24..31}).
// ---------------------------------------------------------------------------
__device__ __forceinline__ v16bf load_frag_a(const __bf16* base, int khalf8) {
  Frag f;
  const uint2* p0 = (const uint2*)(base + khalf8);
  const uint2* p1 = (const uint2*)(base + khalf8 + 16);
  f.u2[0] = p0[0]; f.u2[1] = p0[1];
  f.u2[2] = p1[0]; f.u2[3] = p1[1];
  return f.v;
}
__device__ __forceinline__ v16bf load_frag_b(const __bf16* base, int k16) {
  Frag f;
  const uint2* p = (const uint2*)(base + k16);
  f.u2[0] = p[0]; f.u2[1] = p[1]; f.u2[2] = p[2]; f.u2[3] = p[3];
  return f.v;
}

// ---------------------------------------------------------------------------
// Segment-masked conv-as-GEMM, bf16 hi/lo x3 split, fused bias+ReLU+BN(eval).
// Block = 256 thr = 8 wave32, 64x128 tile; wave = 32x32 patch = 2x2 v8f accs.
// Per K=32 step: 12x V_WMMA_F32_16X16X32_BF16 (hi*hi + hi*lo + lo*hi).
// B tiles staged with GLOBAL_LOAD_ASYNC_TO_LDS_B128 (overlapped with the
// VALU hi/lo conversion of the A tile), completed via s_wait_asynccnt.
// ---------------------------------------------------------------------------
__global__ __launch_bounds__(256)
void segconv_wmma_kernel(const float*  __restrict__ X,    // [Ntok, Din] fp32
                         const int*    __restrict__ seg,  // [Ntok]
                         const __bf16* __restrict__ Wh,   // [K][Dout][DinP]
                         const __bf16* __restrict__ Wl,   // [K][Dout][DinP]
                         const float*  __restrict__ bias,
                         const float*  __restrict__ g,
                         const float*  __restrict__ be,
                         const float*  __restrict__ m,
                         const float*  __restrict__ v,
                         float* __restrict__ H,           // [Ntok, Dout] fp32
                         int Ntok, int Din, int DinP, int Dout) {
  __shared__ __bf16 Ah[BM][LDA];
  __shared__ __bf16 Al[BM][LDA];
  __shared__ __bf16 Bh[BN][LDA];
  __shared__ __bf16 Bl[BN][LDA];

  const int t      = threadIdx.x;
  const int lane   = t & 31;
  const int wid    = t >> 5;
  const int waveM  = wid & 1;            // 2 x 32 rows
  const int waveN  = wid >> 1;           // 4 x 32 cols
  const int lrow   = lane & 15;
  const int khalf8 = (lane >> 4) << 3;   // A: 0 or 8
  const int k16    = (lane >> 4) << 4;   // 0 or 16 (unused)

  const int i0 = blockIdx.x * BM;
  const int j0 = blockIdx.y * BN;

  const v8f zero = {0.f,0.f,0.f,0.f,0.f,0.f,0.f,0.f};
  v8f acc[2][2] = {{zero, zero}, {zero, zero}};

  const int nChunks = DinP / KC;

  for (int tap = 0; tap < KTAPS; ++tap) {
    const int shift = tap - KTAPS / 2;
    for (int ch = 0; ch < nChunks; ++ch) {
      const int d0 = ch * KC;
      const __bf16* baseH = Wh + ((long long)tap * Dout + j0) * DinP + d0;
      const __bf16* baseL = Wl + ((long long)tap * Dout + j0) * DinP + d0;

      // ---- stage B: bf16 weights, global -> LDS (16B lane transfers) ----
#if USE_ASYNC_LDS
      for (int e = t; e < BN * (KC / 8); e += 256) {
        int n  = e >> 3;                 // KC/8 == 8 groups per row
        int k8 = (e & 7) << 3;
        async_copy16(baseH + (long long)n * DinP + k8, &Bh[n][k8]);
        async_copy16(baseL + (long long)n * DinP + k8, &Bl[n][k8]);
      }
#else
      for (int e = t; e < BN * (KC / 8); e += 256) {
        int n  = e >> 3;
        int k8 = (e & 7) << 3;
        *(uint4*)&Bh[n][k8] = *(const uint4*)(baseH + (long long)n * DinP + k8);
        *(uint4*)&Bl[n][k8] = *(const uint4*)(baseL + (long long)n * DinP + k8);
      }
#endif

      // ---- stage A: fp32 -> bf16 hi/lo, segment mask folded in ----
      // (VALU work here overlaps the in-flight async B DMA)
      for (int e = t; e < BM * KC; e += 256) {
        int r  = e >> 6;                 // KC == 64
        int kk = e & (KC - 1);
        int gi  = i0 + r;
        int src = gi + shift;
        int d   = d0 + kk;
        float val = 0.f;
        if (src >= 0 && src < Ntok && d < Din && seg[src] == seg[gi])
          val = X[(long long)src * Din + d];
        __bf16 hi = (__bf16)val;
        Ah[r][kk] = hi;
        Al[r][kk] = (__bf16)(val - (float)hi);
      }

#if USE_ASYNC_LDS
      __builtin_amdgcn_s_wait_asynccnt(0);
#endif
      __syncthreads();

      // ---- 2 K-steps of 32; 12 bf16 WMMAs each ----
      #pragma unroll
      for (int ks = 0; ks < KC; ks += 32) {
        v16bf ah[2], al[2], bh[2], bl[2];
        #pragma unroll
        for (int mi = 0; mi < 2; ++mi) {
          const int row = waveM * 32 + mi * 16 + lrow;
          ah[mi] = load_frag_a(&Ah[row][ks], khalf8);
          al[mi] = load_frag_a(&Al[row][ks], khalf8);
        }
        #pragma unroll
        for (int ni = 0; ni < 2; ++ni) {
          const int n = waveN * 32 + ni * 16 + lrow;
          bh[ni] = load_frag_a(&Bh[n][ks], khalf8);
          bl[ni] = load_frag_a(&Bl[n][ks], khalf8);
        }
        #pragma unroll
        for (int mi = 0; mi < 2; ++mi) {
          #pragma unroll
          for (int ni = 0; ni < 2; ++ni) {
            acc[mi][ni] = __builtin_amdgcn_wmma_f32_16x16x32_bf16(
                false, ah[mi], false, bh[ni], (short)0, acc[mi][ni], false, false);
            acc[mi][ni] = __builtin_amdgcn_wmma_f32_16x16x32_bf16(
                false, ah[mi], false, bl[ni], (short)0, acc[mi][ni], false, false);
            acc[mi][ni] = __builtin_amdgcn_wmma_f32_16x16x32_bf16(
                false, al[mi], false, bh[ni], (short)0, acc[mi][ni], false, false);
          }
        }
      }
      __syncthreads();
    }
  }

  // ---- epilogue: y = relu(x + b) * s + t  (BN eval folded) ----
  // C/D layout: VGPR r -> row = r (+8 for lanes 16..31), col = lane&15.
  const int rtop = i0 + waveM * 32 + ((lane >> 4) << 3);
  #pragma unroll
  for (int mi = 0; mi < 2; ++mi) {
    #pragma unroll
    for (int ni = 0; ni < 2; ++ni) {
      const v8f a = acc[mi][ni];
      const int col = j0 + waveN * 32 + ni * 16 + lrow;
      const float bb = bias[col];
      const float sc = g[col] * rsqrtf(v[col] + BN_EPS);
      const float tt = be[col] - m[col] * sc;
      const int rbase = rtop + mi * 16;
      #pragma unroll
      for (int r = 0; r < 8; ++r) {
        float x = a[r] + bb;
        x = fmaxf(x, 0.f);
        H[(long long)(rbase + r) * Dout + col] = x * sc + tt;
      }
    }
  }
}

// ---------------------------------------------------------------------------
// Final linear (2048 -> 2) + softmax. One wave32 per row; shuffle reduction.
// Output: [logits (N,2) | probs (N,2)] flat.
// ---------------------------------------------------------------------------
__global__ __launch_bounds__(256)
void head_kernel(const float* __restrict__ h,     // [Ntok, D]
                 const float* __restrict__ wlin,  // [2, D]
                 const float* __restrict__ blin,  // [2]
                 float* __restrict__ out,
                 int Ntok, int D) {
  const int wid  = threadIdx.x >> 5;
  const int lane = threadIdx.x & 31;
  const int row  = blockIdx.x * 8 + wid;
  if (row >= Ntok) return;

  float s0 = 0.f, s1 = 0.f;
  for (int d = lane; d < D; d += 32) {
    float hv = h[(long long)row * D + d];
    s0 += hv * wlin[d];
    s1 += hv * wlin[D + d];
  }
  #pragma unroll
  for (int off = 16; off > 0; off >>= 1) {
    s0 += __shfl_down(s0, off, 32);
    s1 += __shfl_down(s1, off, 32);
  }
  if (lane == 0) {
    float l0 = s0 + blin[0];
    float l1 = s1 + blin[1];
    float mm = fmaxf(l0, l1);
    float e0 = __expf(l0 - mm);
    float e1 = __expf(l1 - mm);
    float inv = 1.f / (e0 + e1);
    out[(long long)row * 2 + 0] = l0;
    out[(long long)row * 2 + 1] = l1;
    out[(long long)Ntok * 2 + row * 2 + 0] = e0 * inv;
    out[(long long)Ntok * 2 + row * 2 + 1] = e1 * inv;
  }
}

// ---------------------------------------------------------------------------
extern "C" void kernel_launch(void* const* d_in, const int* in_sizes, int n_in,
                              void* d_out, int out_size, void* d_ws, size_t ws_size,
                              hipStream_t stream) {
  (void)in_sizes; (void)n_in; (void)out_size; (void)ws_size;

  const float* x    = (const float*)d_in[0];
  const int*   seg  = (const int*)  d_in[1];
  const float* w1   = (const float*)d_in[2];
  const float* b1   = (const float*)d_in[3];
  const float* g1   = (const float*)d_in[4];
  const float* be1  = (const float*)d_in[5];
  const float* m1   = (const float*)d_in[6];
  const float* v1   = (const float*)d_in[7];
  const float* w2   = (const float*)d_in[8];
  const float* b2   = (const float*)d_in[9];
  const float* g2   = (const float*)d_in[10];
  const float* be2  = (const float*)d_in[11];
  const float* m2   = (const float*)d_in[12];
  const float* v2   = (const float*)d_in[13];
  const float* wlin = (const float*)d_in[14];
  const float* blin = (const float*)d_in[15];
  float* out = (float*)d_out;

  const int DIN_P1 = 640;   // D_IN  (600) padded to 64
  const int DIN_P2 = 2048;  // D_HID already 64-aligned

  // Workspace (256B-aligned offsets):
  //   wh1/wl1: 9*2048*640*2  = 23,592,960 B each
  //   wh2/wl2: 9*2048*2048*2 = 75,497,472 B each
  //   h1/h2  : 8192*2048*4   = 67,108,864 B each
  char* ws = (char*)d_ws;
  __bf16* wh1 = (__bf16*)(ws);
  __bf16* wl1 = (__bf16*)(ws + 23592960LL);
  __bf16* wh2 = (__bf16*)(ws + 47185920LL);
  __bf16* wl2 = (__bf16*)(ws + 122683392LL);
  float*  h1  = (float*) (ws + 198180864LL);
  float*  h2  = (float*) (ws + 265289728LL);

  {
    long long tot = (long long)D_HID * DIN_P1 * KTAPS;
    split_transpose_kernel<<<(int)((tot + 255) / 256), 256, 0, stream>>>(
        w1, wh1, wl1, D_HID, D_IN, DIN_P1, KTAPS);
  }
  {
    long long tot = (long long)D_HID * DIN_P2 * KTAPS;
    split_transpose_kernel<<<(int)((tot + 255) / 256), 256, 0, stream>>>(
        w2, wh2, wl2, D_HID, D_HID, DIN_P2, KTAPS);
  }

  dim3 grid(N_TOK / BM, D_HID / BN);
  segconv_wmma_kernel<<<grid, 256, 0, stream>>>(
      x,  seg, wh1, wl1, b1, g1, be1, m1, v1, h1, N_TOK, D_IN,  DIN_P1, D_HID);
  segconv_wmma_kernel<<<grid, 256, 0, stream>>>(
      h1, seg, wh2, wl2, b2, g2, be2, m2, v2, h2, N_TOK, D_HID, DIN_P2, D_HID);

  head_kernel<<<N_TOK / 8, 256, 0, stream>>>(h2, wlin, blin, out, N_TOK, D_HID);
}
